// QwenAttentionLayer_88390426952209
// MI455X (gfx1250) — matrix operand order repaired and checked by reading, not compiled, not run on hardware
//
#include <hip/hip_runtime.h>

// ---------------------------------------------------------------------------
// CDNA5 (gfx1250) wave32 WMMA implementation of a Qwen-style transformer layer
// with async global->LDS double buffering (ASYNCcnt path).
// ---------------------------------------------------------------------------

typedef __attribute__((ext_vector_type(16))) __bf16 v16bf;
typedef __attribute__((ext_vector_type(8)))  float  v8f;

union FragBF {
  v16bf v;
  uint4 u[2];
};

__device__ __forceinline__ unsigned short f2bf(float f) {
  unsigned int u = __float_as_uint(f);
  u += 0x7FFFu + ((u >> 16) & 1u);          // round-to-nearest-even
  return (unsigned short)(u >> 16);
}
__device__ __forceinline__ float bf2f(unsigned short s) {
  return __uint_as_float(((unsigned int)s) << 16);
}

// ---- CDNA5 async global->LDS copy (16B per lane), tracked by ASYNCcnt ------
__device__ __forceinline__ void async_copy16(unsigned lds_off, const void* gptr) {
  asm volatile("global_load_async_to_lds_b128 %0, %1, off"
               :: "v"(lds_off), "v"((unsigned long long)(uintptr_t)gptr)
               : "memory");
}
__device__ __forceinline__ void wait_async_0() {
  asm volatile("s_wait_asynccnt 0x0" ::: "memory");
}
__device__ __forceinline__ void wait_async_8() {
  asm volatile("s_wait_asynccnt 0x8" ::: "memory");
}

// ---------------------------------------------------------------------------
// Generic bf16 WMMA GEMM:  C[M,N] = A[M,K] * Bt[N,K]^T  (Bt is N-major)
// Block tile 128x128, K-step 64, 8 waves (2x4), each wave 64x32 (4x2 tiles).
// Double-buffered async global->LDS staging.
// mode 0: Cb = bf16(acc)   mode 1: Cf = acc   mode 2: Cf = acc + R
// ---------------------------------------------------------------------------
#define GBM 128
#define GBN 128
#define GBK 64
#define GLDT 72                 // 64 + 8 pad (144B rows, 16B aligned)
#define GBUFE (GBM * GLDT)      // elements per tile buffer
#define GEMM_SMEM_BYTES (4 * GBUFE * 2)   // As0,Bs0,As1,Bs1 = 73728 B

__global__ __launch_bounds__(256) void k_gemm_bf16(
    const unsigned short* __restrict__ A, int lda,
    const unsigned short* __restrict__ Bt, int ldb,
    int M, int N, int K,
    float* __restrict__ Cf, unsigned short* __restrict__ Cb,
    const float* __restrict__ R, int ldc, int mode)
{
  extern __shared__ unsigned short smem[];
  unsigned short* As0 = smem;
  unsigned short* Bs0 = smem + GBUFE;
  unsigned short* As1 = smem + 2 * GBUFE;
  unsigned short* Bs1 = smem + 3 * GBUFE;

  const int t    = threadIdx.x;
  const int wave = t >> 5, lane = t & 31;
  const int wrow = wave >> 2, wcol = wave & 3;      // 2 x 4 wave grid
  const int half = lane >> 4, l16 = lane & 15;
  const int bm = blockIdx.y * GBM, bn = blockIdx.x * GBN;
  (void)M; (void)N;

  const v8f zero8 = {0.f,0.f,0.f,0.f,0.f,0.f,0.f,0.f};
  v8f acc[4][2];
  #pragma unroll
  for (int i = 0; i < 4; i++)
    #pragma unroll
    for (int j = 0; j < 2; j++) acc[i][j] = zero8;

  auto issue = [&](int k0, unsigned short* Ad, unsigned short* Bd) {
    #pragma unroll
    for (int i = 0; i < 4; i++) {
      int c = t + 256 * i;                 // 1024 chunks of 8 bf16
      int row = c >> 3, col8 = c & 7;
      async_copy16((unsigned)(uintptr_t)(Ad + row * GLDT + col8 * 8),
                   &A[(size_t)(bm + row) * lda + k0 + col8 * 8]);
    }
    #pragma unroll
    for (int i = 0; i < 4; i++) {
      int c = t + 256 * i;
      int row = c >> 3, col8 = c & 7;
      async_copy16((unsigned)(uintptr_t)(Bd + row * GLDT + col8 * 8),
                   &Bt[(size_t)(bn + row) * ldb + k0 + col8 * 8]);
    }
  };

  const int niter = K / GBK;
  issue(0, As0, Bs0);

  for (int it = 0; it < niter; it++) {
    unsigned short* Ac = (it & 1) ? As1 : As0;
    unsigned short* Bc = (it & 1) ? Bs1 : Bs0;
    if (it + 1 < niter) {
      issue((it + 1) * GBK, (it & 1) ? As0 : As1, (it & 1) ? Bs0 : Bs1);
      wait_async_8();                      // retire current tile's 8 copies
    } else {
      wait_async_0();
    }
    if (it + 2 < niter) {                  // warm L2 for tile after next
      __builtin_prefetch(&A[(size_t)(bm + (t >> 1)) * lda + (it + 2) * GBK], 0, 1);
      __builtin_prefetch(&Bt[(size_t)(bn + (t >> 1)) * ldb + (it + 2) * GBK], 0, 1);
    }
    __syncthreads();

    #pragma unroll
    for (int kc = 0; kc < GBK; kc += 32) {
      FragBF a[4], b[2];
      #pragma unroll
      for (int mt = 0; mt < 4; mt++) {
        // A layout: lane row = l16; lanes0-15 K=kc+0..7 & kc+16..23,
        //           lanes16-31 K=kc+8..15 & kc+24..31
        const unsigned short* p =
            &Ac[(wrow * 64 + mt * 16 + l16) * GLDT + kc + half * 8];
        a[mt].u[0] = *(const uint4*)p;
        a[mt].u[1] = *(const uint4*)(p + 16);
      }
      #pragma unroll
      for (int nt = 0; nt < 2; nt++) {
        // B layout: lane col = l16; lanes0-15 K=kc+0..7 & kc+16..23, lanes16-31 K=kc+8..15 & kc+24..31
        const unsigned short* p =
            &Bc[(wcol * 32 + nt * 16 + l16) * GLDT + kc + half * 8];
        b[nt].u[0] = *(const uint4*)p;
        b[nt].u[1] = *(const uint4*)(p + 16);
      }
      #pragma unroll
      for (int mt = 0; mt < 4; mt++)
        #pragma unroll
        for (int nt = 0; nt < 2; nt++)
          acc[mt][nt] = __builtin_amdgcn_wmma_f32_16x16x32_bf16(
              false, a[mt].v, false, b[nt].v,
              (short)0, acc[mt][nt], false, false);
    }
    __syncthreads();
  }

  // ---- epilogue: C layout lane col = l16, VGPR r = row (r / r+8 by half) ----
  const int row0 = bm + wrow * 64;
  const int col0 = bn + wcol * 32;
  #pragma unroll
  for (int mt = 0; mt < 4; mt++)
    #pragma unroll
    for (int nt = 0; nt < 2; nt++)
      #pragma unroll
      for (int r = 0; r < 8; r++) {
        int row = row0 + mt * 16 + half * 8 + r;
        int col = col0 + nt * 16 + l16;
        size_t o = (size_t)row * ldc + col;
        float v = acc[mt][nt][r];
        if (mode == 0)      Cb[o] = f2bf(v);
        else if (mode == 1) Cf[o] = v;
        else                Cf[o] = v + R[o];
      }
}

// ---------------------------------------------------------------------------
// Flash attention (causal, GQA). One block = one (b,h) x 128 query rows.
// Each wave owns 16 query rows -> row softmax = 16-lane shfl reductions.
// Q*K^T and P*V both via v_wmma_f32_16x16x32_bf16, fp32 online softmax.
// K tile staged with async global->LDS; V tile transposed synchronously.
// ---------------------------------------------------------------------------
#define FA_BQ 128
#define FA_BK 64

__global__ __launch_bounds__(256) void k_flash_attn(
    const unsigned short* __restrict__ Q,   // (B, H, L, 128)
    const unsigned short* __restrict__ Kk,  // (B, HKV, L, 128)
    const unsigned short* __restrict__ V,   // (B, HKV, L, 128)
    unsigned short* __restrict__ O,         // (B, L, H*128)
    int Bn, int Hn, int HKVn, int Ln, float scale)
{
  const int DHc = 128;
  __shared__ __align__(16) unsigned short Kt[64 * 136];   // keys x DH
  __shared__ __align__(16) unsigned short Vt[128 * 72];   // DH x keys (transposed)
  __shared__ __align__(16) unsigned short Pl[8 * 16 * 72];// per-wave P slab

  const int t    = threadIdx.x;
  const int wave = t >> 5, lane = t & 31;
  const int half = lane >> 4, l16 = lane & 15;
  const int qblk = blockIdx.x;
  const int bh   = blockIdx.y;
  const int b = bh / Hn, h = bh % Hn;
  const int hkv = h / (Hn / HKVn);

  const unsigned short* Qp = Q  + ((size_t)(b * Hn + h) * Ln + qblk * FA_BQ) * DHc;
  const unsigned short* Kp = Kk + (size_t)(b * HKVn + hkv) * Ln * DHc;
  const unsigned short* Vp = V  + (size_t)(b * HKVn + hkv) * Ln * DHc;

  // Q fragments for this wave's 16 rows, 4 K-chunks of 32 (DH=128)
  FragBF qf[4];
  {
    const unsigned short* qrow = Qp + (size_t)(wave * 16 + l16) * DHc;
    #pragma unroll
    for (int kc = 0; kc < 4; kc++) {
      qf[kc].u[0] = *(const uint4*)(qrow + kc * 32 + half * 8);
      qf[kc].u[1] = *(const uint4*)(qrow + kc * 32 + half * 8 + 16);
    }
  }

  const v8f zero8 = {0.f,0.f,0.f,0.f,0.f,0.f,0.f,0.f};
  v8f o[8];
  #pragma unroll
  for (int i = 0; i < 8; i++) o[i] = zero8;
  float mrow[8], lrow[8];
  #pragma unroll
  for (int r = 0; r < 8; r++) { mrow[r] = -3.0e38f; lrow[r] = 0.f; }

  const int jmax = 2 * qblk + 1;           // causal: key blocks up to diagonal
  for (int j = 0; j <= jmax; j++) {
    // ---- stage K tile (64 x 128) via async copies ----
    #pragma unroll
    for (int i = 0; i < 4; i++) {
      int c = t + 256 * i;
      int key = c >> 4, d8 = c & 15;
      async_copy16((unsigned)(uintptr_t)&Kt[key * 136 + d8 * 8],
                   &Kp[(size_t)(j * FA_BK + key) * DHc + d8 * 8]);
    }
    // ---- V tile transposed (128 x 64), overlaps with async K copies ----
    #pragma unroll
    for (int i = 0; i < 4; i++) {
      int c = t + 256 * i;
      int key = c >> 4, d8 = c & 15;
      uint4 raw = *(const uint4*)&Vp[(size_t)(j * FA_BK + key) * DHc + d8 * 8];
      const unsigned short* e = (const unsigned short*)&raw;
      #pragma unroll
      for (int q2 = 0; q2 < 8; q2++)
        Vt[(d8 * 8 + q2) * 72 + key] = e[q2];
    }
    wait_async_0();
    __syncthreads();

    // ---- S = Q K^T : this wave's 16 rows x 64 keys ----
    v8f s[4];
    #pragma unroll
    for (int nt = 0; nt < 4; nt++) {
      s[nt] = zero8;
      #pragma unroll
      for (int kc = 0; kc < 4; kc++) {
        FragBF kf;
        const unsigned short* p = &Kt[(nt * 16 + l16) * 136 + kc * 32 + half * 8];
        kf.u[0] = *(const uint4*)p;
        kf.u[1] = *(const uint4*)(p + 16);
        s[nt] = __builtin_amdgcn_wmma_f32_16x16x32_bf16(
            false, qf[kc].v, false, kf.v, (short)0, s[nt], false, false);
      }
    }

    // ---- scale + causal mask ----
    const int qi0 = qblk * FA_BQ + wave * 16 + half * 8;
    #pragma unroll
    for (int nt = 0; nt < 4; nt++) {
      int kj = j * FA_BK + nt * 16 + l16;
      #pragma unroll
      for (int r = 0; r < 8; r++) {
        float v = s[nt][r] * scale;
        s[nt][r] = (kj > qi0 + r) ? -1.0e30f : v;
      }
    }

    // ---- online softmax (per-row, 16-lane shfl reductions) ----
    #pragma unroll
    for (int r = 0; r < 8; r++) {
      float x = fmaxf(fmaxf(s[0][r], s[1][r]), fmaxf(s[2][r], s[3][r]));
      #pragma unroll
      for (int m2 = 1; m2 < 16; m2 <<= 1) x = fmaxf(x, __shfl_xor(x, m2, 32));
      float mnew = fmaxf(mrow[r], x);
      float sc = __expf(mrow[r] - mnew);
      float psum = 0.f;
      #pragma unroll
      for (int nt = 0; nt < 4; nt++) {
        float p = __expf(s[nt][r] - mnew);
        s[nt][r] = p;
        psum += p;
      }
      #pragma unroll
      for (int m2 = 1; m2 < 16; m2 <<= 1) psum += __shfl_xor(psum, m2, 32);
      lrow[r] = lrow[r] * sc + psum;
      mrow[r] = mnew;
      #pragma unroll
      for (int ot = 0; ot < 8; ot++) o[ot][r] *= sc;
    }

    // ---- P (C-layout) -> LDS -> A-layout fragments ----
    unsigned short* pw = &Pl[wave * 16 * 72];
    #pragma unroll
    for (int nt = 0; nt < 4; nt++)
      #pragma unroll
      for (int r = 0; r < 8; r++)
        pw[(r + half * 8) * 72 + nt * 16 + l16] = f2bf(s[nt][r]);
    __syncthreads();

    // ---- O += P V ----
    #pragma unroll
    for (int kc = 0; kc < 2; kc++) {
      FragBF pf;
      const unsigned short* pp = &pw[l16 * 72 + kc * 32 + half * 8];
      pf.u[0] = *(const uint4*)pp;
      pf.u[1] = *(const uint4*)(pp + 16);
      #pragma unroll
      for (int ot = 0; ot < 8; ot++) {
        FragBF vf;
        const unsigned short* vp = &Vt[(ot * 16 + l16) * 72 + kc * 32 + half * 8];
        vf.u[0] = *(const uint4*)vp;
        vf.u[1] = *(const uint4*)(vp + 16);
        o[ot] = __builtin_amdgcn_wmma_f32_16x16x32_bf16(
            false, pf.v, false, vf.v, (short)0, o[ot], false, false);
      }
    }
    __syncthreads();
  }

  // ---- normalize, write (B, L, H*DH) bf16 ----
  const int HD = Hn * DHc;
  #pragma unroll
  for (int r = 0; r < 8; r++) {
    float inv = 1.f / lrow[r];
    int row = qblk * FA_BQ + wave * 16 + half * 8 + r;
    size_t base = ((size_t)b * Ln + row) * HD + h * DHc;
    #pragma unroll
    for (int ot = 0; ot < 8; ot++)
      O[base + ot * 16 + l16] = f2bf(o[ot][r] * inv);
  }
}

// ---------------------------------------------------------------------------
// RMSNorm (fp32 in) -> bf16 out. One block per row.
// ---------------------------------------------------------------------------
__global__ __launch_bounds__(256) void k_rmsnorm_bf16(
    const float* __restrict__ x, const float* __restrict__ w,
    unsigned short* __restrict__ out, int Dn)
{
  __shared__ float red[256];
  const int row = blockIdx.x;
  const int t = threadIdx.x;
  const float* xr = x + (size_t)row * Dn;
  float s = 0.f;
  for (int i = t; i < Dn; i += 256) { float v = xr[i]; s += v * v; }
  red[t] = s;
  __syncthreads();
  for (int off = 128; off > 0; off >>= 1) {
    if (t < off) red[t] += red[t + off];
    __syncthreads();
  }
  float inv = rsqrtf(red[0] / (float)Dn + 1e-6f);
  unsigned short* orow = out + (size_t)row * Dn;
  for (int i = t; i < Dn; i += 256) orow[i] = f2bf(xr[i] * inv * w[i]);
}

// ---------------------------------------------------------------------------
// Transpose + fp32->bf16 + zero-pad:  src (K,N) f32  ->  dst (Np,Kp) bf16
// ---------------------------------------------------------------------------
__global__ __launch_bounds__(256) void k_transpose_bf16(
    const float* __restrict__ src, unsigned short* __restrict__ dst,
    int K, int N, int Kp, int Np)
{
  __shared__ float tile[32][33];
  const int tx = threadIdx.x & 31, ty = threadIdx.x >> 5;
  const int k0 = blockIdx.y * 32, n0 = blockIdx.x * 32;
  #pragma unroll
  for (int i = 0; i < 4; i++) {
    int k = k0 + ty + i * 8, n = n0 + tx;
    tile[ty + i * 8][tx] = (k < K && n < N) ? src[(size_t)k * N + n] : 0.f;
  }
  __syncthreads();
  #pragma unroll
  for (int i = 0; i < 4; i++) {
    int n = n0 + ty + i * 8, k = k0 + tx;
    if (n < Np && k < Kp) dst[(size_t)n * Kp + k] = f2bf(tile[tx][ty + i * 8]);
  }
}

// ---------------------------------------------------------------------------
// RoPE + head-major repack: (B, L, nh, 128) -> (B, nh, L, 128), bf16
// ---------------------------------------------------------------------------
__global__ __launch_bounds__(256) void k_rope_pack(
    const unsigned short* __restrict__ in, unsigned short* __restrict__ out,
    int Bn, int nh, int Ln, int doRope)
{
  const int DHc = 128, HDH = 64;
  int idx = blockIdx.x * blockDim.x + threadIdx.x;
  int total = Bn * nh * Ln * HDH;
  if (idx >= total) return;
  int d  = idx % HDH;
  int l  = (idx / HDH) % Ln;
  int hh = (idx / (HDH * Ln)) % nh;
  int b  = idx / (HDH * Ln * nh);
  size_t ib = (((size_t)b * Ln + l) * nh + hh) * DHc;
  float x1 = bf2f(in[ib + d]);
  float x2 = bf2f(in[ib + d + HDH]);
  float o1 = x1, o2 = x2;
  if (doRope) {
    float ang = (float)l * __powf(10000.f, -(float)(2 * d) * (1.f / 128.f));
    float sn, cs;
    __sincosf(ang, &sn, &cs);
    o1 = x1 * cs - x2 * sn;
    o2 = x2 * cs + x1 * sn;
  }
  size_t ob = (((size_t)b * nh + hh) * Ln + l) * DHc;
  out[ob + d]       = f2bf(o1);
  out[ob + d + HDH] = f2bf(o2);
}

// ---------------------------------------------------------------------------
// SwiGLU elementwise: Hff = bf16( silu(G) * U )
// ---------------------------------------------------------------------------
__global__ __launch_bounds__(256) void k_silu_mul(
    const unsigned short* __restrict__ G, const unsigned short* __restrict__ U,
    unsigned short* __restrict__ Hf, int total)
{
  int idx = blockIdx.x * blockDim.x + threadIdx.x;
  if (idx >= total) return;
  float g = bf2f(G[idx]);
  float u = bf2f(U[idx]);
  float s = g / (1.f + __expf(-g));
  Hf[idx] = f2bf(s * u);
}

// ---------------------------------------------------------------------------
// Launch
// ---------------------------------------------------------------------------
extern "C" void kernel_launch(void* const* d_in, const int* in_sizes, int n_in,
                              void* d_out, int out_size, void* d_ws, size_t ws_size,
                              hipStream_t stream) {
  (void)in_sizes; (void)n_in; (void)out_size; (void)ws_size;
  const int Bc = 2, Lc = 2048, Dc = 2048, Hc = 16, HKVc = 4, FFc = 5461, FFP = 5504;
  const int Mc = Bc * Lc;                              // 4096 token rows
  const float scale = 0.08838834764831845f;            // 1/sqrt(128)

  const float* x   = (const float*)d_in[0];
  const float* wn1 = (const float*)d_in[1];
  const float* wq  = (const float*)d_in[2];
  const float* wk  = (const float*)d_in[3];
  const float* wv  = (const float*)d_in[4];
  const float* wo  = (const float*)d_in[5];
  const float* wn2 = (const float*)d_in[6];
  const float* wg  = (const float*)d_in[7];
  const float* wu  = (const float*)d_in[8];
  const float* wd  = (const float*)d_in[9];
  float* out = (float*)d_out;

  char* ws = (char*)d_ws;
  size_t off = 0;
  auto alloc = [&](size_t bytes) -> char* {
    char* p = ws + off;
    off += (bytes + 255) & ~(size_t)255;
    return p;
  };
  unsigned short* hbf  = (unsigned short*)alloc((size_t)Mc * Dc * 2);
  unsigned short* wq_t = (unsigned short*)alloc((size_t)2048 * 2048 * 2);
  unsigned short* wk_t = (unsigned short*)alloc((size_t)512 * 2048 * 2);
  unsigned short* wv_t = (unsigned short*)alloc((size_t)512 * 2048 * 2);
  unsigned short* wo_t = (unsigned short*)alloc((size_t)2048 * 2048 * 2);
  unsigned short* wg_t = (unsigned short*)alloc((size_t)FFP * 2048 * 2);
  unsigned short* wu_t = (unsigned short*)alloc((size_t)FFP * 2048 * 2);
  unsigned short* wd_t = (unsigned short*)alloc((size_t)2048 * FFP * 2);
  unsigned short* qlin = (unsigned short*)alloc((size_t)Mc * 2048 * 2);
  unsigned short* klin = (unsigned short*)alloc((size_t)Mc * 512 * 2);
  unsigned short* vlin = (unsigned short*)alloc((size_t)Mc * 512 * 2);
  unsigned short* Qr   = (unsigned short*)alloc((size_t)Mc * 2048 * 2);
  unsigned short* Kr   = (unsigned short*)alloc((size_t)Mc * 512 * 2);
  unsigned short* Vr   = (unsigned short*)alloc((size_t)Mc * 512 * 2);
  unsigned short* Ao   = (unsigned short*)alloc((size_t)Mc * 2048 * 2);
  float*          xmid = (float*)alloc((size_t)Mc * 2048 * 4);
  unsigned short* Gb   = (unsigned short*)alloc((size_t)Mc * FFP * 2);
  unsigned short* Ub   = (unsigned short*)alloc((size_t)Mc * FFP * 2);
  unsigned short* Hff  = (unsigned short*)alloc((size_t)Mc * FFP * 2);

  dim3 blk(256);
  auto tgrid = [](int Np, int Kp) { return dim3((Np + 31) / 32, (Kp + 31) / 32); };
  const size_t gsm = GEMM_SMEM_BYTES;

  // ---- weights: fp32 (K,N) -> bf16 (N,K) transposed, FF dims padded ----
  k_transpose_bf16<<<tgrid(2048, 2048), blk, 0, stream>>>(wq, wq_t, 2048, 2048, 2048, 2048);
  k_transpose_bf16<<<tgrid(512,  2048), blk, 0, stream>>>(wk, wk_t, 2048,  512, 2048,  512);
  k_transpose_bf16<<<tgrid(512,  2048), blk, 0, stream>>>(wv, wv_t, 2048,  512, 2048,  512);
  k_transpose_bf16<<<tgrid(2048, 2048), blk, 0, stream>>>(wo, wo_t, 2048, 2048, 2048, 2048);
  k_transpose_bf16<<<tgrid(FFP,  2048), blk, 0, stream>>>(wg, wg_t, 2048,  FFc, 2048,  FFP);
  k_transpose_bf16<<<tgrid(FFP,  2048), blk, 0, stream>>>(wu, wu_t, 2048,  FFc, 2048,  FFP);
  k_transpose_bf16<<<tgrid(2048,  FFP), blk, 0, stream>>>(wd, wd_t,  FFc, 2048,  FFP, 2048);

  // ---- attention half ----
  k_rmsnorm_bf16<<<Mc, blk, 0, stream>>>(x, wn1, hbf, Dc);

  k_gemm_bf16<<<dim3(2048/128, Mc/128), blk, gsm, stream>>>(hbf, Dc, wq_t, Dc, Mc, 2048, Dc, nullptr, qlin, nullptr, 2048, 0);
  k_gemm_bf16<<<dim3( 512/128, Mc/128), blk, gsm, stream>>>(hbf, Dc, wk_t, Dc, Mc,  512, Dc, nullptr, klin, nullptr,  512, 0);
  k_gemm_bf16<<<dim3( 512/128, Mc/128), blk, gsm, stream>>>(hbf, Dc, wv_t, Dc, Mc,  512, Dc, nullptr, vlin, nullptr,  512, 0);

  int tot_q  = Bc * Hc   * Lc * 64;
  int tot_kv = Bc * HKVc * Lc * 64;
  k_rope_pack<<<(tot_q  + 255)/256, blk, 0, stream>>>(qlin, Qr, Bc, Hc,   Lc, 1);
  k_rope_pack<<<(tot_kv + 255)/256, blk, 0, stream>>>(klin, Kr, Bc, HKVc, Lc, 1);
  k_rope_pack<<<(tot_kv + 255)/256, blk, 0, stream>>>(vlin, Vr, Bc, HKVc, Lc, 0);

  k_flash_attn<<<dim3(Lc/128, Bc*Hc), blk, 0, stream>>>(Qr, Kr, Vr, Ao, Bc, Hc, HKVc, Lc, scale);

  k_gemm_bf16<<<dim3(2048/128, Mc/128), blk, gsm, stream>>>(Ao, 2048, wo_t, 2048, Mc, 2048, 2048, xmid, nullptr, x, 2048, 2);

  // ---- FFN half ----
  k_rmsnorm_bf16<<<Mc, blk, 0, stream>>>(xmid, wn2, hbf, Dc);

  k_gemm_bf16<<<dim3(FFP/128, Mc/128), blk, gsm, stream>>>(hbf, Dc, wg_t, Dc, Mc, FFP, Dc, nullptr, Gb, nullptr, FFP, 0);
  k_gemm_bf16<<<dim3(FFP/128, Mc/128), blk, gsm, stream>>>(hbf, Dc, wu_t, Dc, Mc, FFP, Dc, nullptr, Ub, nullptr, FFP, 0);

  int tot_ff = Mc * FFP;
  k_silu_mul<<<(tot_ff + 255)/256, blk, 0, stream>>>(Gb, Ub, Hff, tot_ff);

  k_gemm_bf16<<<dim3(2048/128, Mc/128), blk, gsm, stream>>>(Hff, FFP, wd_t, FFP, Mc, 2048, FFP, out, nullptr, xmid, 2048, 2);
}
